// Aiteration_62843961475391
// MI455X (gfx1250) — compile-verified, matrix operand order checked by reading, not run on hardware
//
// CT learned-iteration pipeline for gfx1250 (MI455X), wave32 + WMMA bf16.
// Heavy ops (5x5 64->64 convs, DFT-as-GEMM, filter-as-GEMM) use
// v_wmma_f32_16x16x32_bf16 with fully padded GEMM operands (conv A fragment
// loads are guarded); spmv gathers stay VALU with B128 index/value streaming.
#include <hip/hip_runtime.h>
#include <cstddef>

#define CDIV(a, b) (((a) + (b)-1) / (b))

namespace {
constexpr int kNAng = 360;
constexpr int kLA   = 361;
constexpr int kLB   = 180;
constexpr int kH    = 256;
constexpr int kM    = kNAng * kLA;   // 129960 sinogram elems
constexpr int kNPix = kH * kH;       // 65536
constexpr int kCRows = kLB * kLA;    // 64980 (= (NANG-LL)*LA too)
// padded GEMM geometry: all GEMMs are [360,361]x[361,361]
constexpr int kKp = 384;             // K padded to 12 chunks of 32
constexpr int kRp = 368;             // rows padded to 23 tiles of 16 (A rows & Bt rows)
constexpr int kPadElems = kRp * kKp; // 141312
}

typedef __attribute__((ext_vector_type(16))) __bf16         v16bf;
typedef __attribute__((ext_vector_type(8)))  float          v8f;
typedef __attribute__((ext_vector_type(16))) unsigned short v16us;
typedef __attribute__((ext_vector_type(8)))  unsigned short v8us;

__device__ __forceinline__ unsigned short f2bf(float f) {
  unsigned int u = __builtin_bit_cast(unsigned int, f);
  u += 0x7fffu + ((u >> 16) & 1u);           // round-to-nearest-even
  return (unsigned short)(u >> 16);
}
__device__ __forceinline__ float bf2f(unsigned short s) {
  unsigned int u = ((unsigned int)s) << 16;
  return __builtin_bit_cast(float, u);
}

// ---------------- elementwise / prep kernels ----------------

// f32 [360*361] -> zero-padded bf16 [368][384]
__global__ void cvt_pad_kernel(const float* __restrict__ a,
                               unsigned short* __restrict__ b) {
  int t = blockIdx.x * blockDim.x + threadIdx.x;
  if (t >= kPadElems) return;
  int col = t % kKp, row = t / kKp;
  unsigned short v = 0;
  if (row < kNAng && col < kLA) v = f2bf(a[row * kLA + col]);
  b[t] = v;
}

// (5,5,Cin,Cout) f32 -> [tap][cout][cin] bf16
__global__ void wtrans_kernel(const float* __restrict__ w,
                              unsigned short* __restrict__ wT, int Cin, int Cout) {
  int t = blockIdx.x * blockDim.x + threadIdx.x;
  int total = 25 * Cin * Cout;
  if (t >= total) return;
  int ci = t % Cin;
  int rest = t / Cin;
  int co = rest % Cout;
  int tap = rest / Cout;
  wT[t] = f2bf(w[(tap * Cin + ci) * Cout + co]);
}

__global__ void bn_prep_kernel(const float* __restrict__ gmm, const float* __restrict__ bet,
                               const float* __restrict__ mean, const float* __restrict__ var,
                               float* __restrict__ sc, float* __restrict__ sh) {
  int c = threadIdx.x;  // 64
  float s = gmm[c] * rsqrtf(var[c] + 1e-3f);
  sc[c] = s;
  sh[c] = bet[c] - mean[c] * s;
}

// DFT bases in padded Bt layout [368 rows (out col)][384 (K)]; symmetric matrix.
__global__ void dft_basis_kernel(unsigned short* __restrict__ cosB,
                                 unsigned short* __restrict__ sinB) {
  int t = blockIdx.x * blockDim.x + threadIdx.x;
  if (t >= kPadElems) return;
  int i = t % kKp, j = t / kKp;
  unsigned short c = 0, s = 0;
  if (i < kLA && j < kLA) {
    int r = (int)(((long long)i * (long long)j) % kLA);  // exact phase reduction
    float arg = 6.2831853071795864f * (float)r / (float)kLA;
    c = f2bf(cosf(arg));
    s = f2bf(sinf(arg));
  }
  cosB[t] = c;
  sinB[t] = s;
}

// Filter as padded Bt [368 (j out)][384 (n in)]: filt[j] = sum_t s1[j+t-180]*w_c[t]
//   => Bt[j][n] = w_c[n - j + 180]
__global__ void filt_basis_kernel(const float* __restrict__ w_c,
                                  unsigned short* __restrict__ fB) {
  int t = blockIdx.x * blockDim.x + threadIdx.x;
  if (t >= kPadElems) return;
  int n = t % kKp, j = t / kKp;
  unsigned short v = 0;
  if (n < kLA && j < kLA) {
    int d = n - j + 180;
    if (d >= 0 && d < kLA) v = f2bf(w_c[d]);
  }
  fB[t] = v;
}

// fixed-nnz sparse matrix-vector: y[r] = sum_k val[r,k]*x[idx[r,k]]; K % 4 == 0
__global__ void spmv_kernel(const int* __restrict__ idx, const float* __restrict__ val,
                            const float* __restrict__ x, float* __restrict__ y,
                            int R, int K) {
  int r = blockIdx.x * blockDim.x + threadIdx.x;
  if (r >= R) return;
  const int* ip = idx + (size_t)r * K;
  const float* vp = val + (size_t)r * K;
  float s = 0.f;
  for (int k = 0; k < K; k += 4) {
    int4 i4 = *(const int4*)(ip + k);
    float4 v4 = *(const float4*)(vp + k);
    s += v4.x * x[i4.x] + v4.y * x[i4.y] + v4.z * x[i4.z] + v4.w * x[i4.w];
  }
  y[r] = s;
}

__global__ void sub_kernel(const float* __restrict__ a, const float* __restrict__ b,
                           float* __restrict__ o, int n) {
  int t = blockIdx.x * blockDim.x + threadIdx.x;
  if (t < n) o[t] = a[t] - b[t];
}

__global__ void combo_kernel(const float* __restrict__ a1, const float* __restrict__ Af,
                             const float* __restrict__ z, const float* __restrict__ l1,
                             const float* __restrict__ l2, float* __restrict__ o, int n) {
  int t = blockIdx.x * blockDim.x + threadIdx.x;
  if (t < n) o[t] = l1[0] * a1[t] + l2[0] * (Af[t] - z[t]);
}

// s1[a,j] = h * combo[a,j] * cos(alpha[j]) -> padded bf16 [368][384]
__global__ void s1_pad_kernel(const float* __restrict__ combo,
                              const float* __restrict__ alpha,
                              unsigned short* __restrict__ o) {
  int t = blockIdx.x * blockDim.x + threadIdx.x;
  if (t >= kPadElems) return;
  int col = t % kKp, row = t / kKp;
  unsigned short v = 0;
  if (row < kNAng && col < kLA) {
    float h = alpha[1] - alpha[0];
    v = f2bf(h * combo[row * kLA + col] * cosf(alpha[col]));
  }
  o[t] = v;
}

__global__ void updatef_kernel(const float* __restrict__ f0, const float* __restrict__ a3,
                               const float* __restrict__ u, const float* __restrict__ l3,
                               const float* __restrict__ l4, float* __restrict__ f1, int n) {
  int t = blockIdx.x * blockDim.x + threadIdx.x;
  if (t < n) f1[t] = l4[0] * f0[t] - a3[t] + l3[0] * u[t];
}

// ---------------- WMMA kernels ----------------

// C[M,N] = alpha * A @ Bt^T (+C if acc). A,Bt are zero-padded [368][384] bf16,
// so all fragment loads are unconditional B128s; only C stores are guarded.
__global__ void gemm_bf16_wmma_kernel(const unsigned short* __restrict__ A,
                                      const unsigned short* __restrict__ Bt,
                                      float* __restrict__ C, int M, int N,
                                      float alpha, int acc_flag) {
  int m0 = blockIdx.x << 4, n0 = blockIdx.y << 4;
  int lane = threadIdx.x;
  int half = lane >> 4, q = lane & 15;
  const unsigned short* ap = A + (size_t)(m0 + q) * kKp;
  const unsigned short* bp = Bt + (size_t)(n0 + q) * kKp;
  v8f acc;
#pragma unroll
  for (int r = 0; r < 8; ++r) acc[r] = 0.f;
#pragma unroll 4
  for (int kc = 0; kc < kKp; kc += 32) {
    int kb = kc + half * 8;
    v8us alo = *(const v8us*)(ap + kb);
    v8us ahi = *(const v8us*)(ap + kb + 16);
    v8us blo = *(const v8us*)(bp + kb);
    v8us bhi = *(const v8us*)(bp + kb + 16);
    v16us au, bu;
#pragma unroll
    for (int j = 0; j < 8; ++j) {
      au[j] = alo[j]; au[j + 8] = ahi[j];
      bu[j] = blo[j]; bu[j + 8] = bhi[j];
    }
    acc = __builtin_amdgcn_wmma_f32_16x16x32_bf16(
        false, __builtin_bit_cast(v16bf, au),
        false, __builtin_bit_cast(v16bf, bu),
        (short)0, acc, false, false);
  }
  int cw = n0 + q;
  if (cw < N) {
#pragma unroll
    for (int r = 0; r < 8; ++r) {
      int rm = m0 + half * 8 + r;
      if (rm < M) {
        float v = alpha * acc[r];
        if (acc_flag) C[(size_t)rm * N + cw] += v;
        else C[(size_t)rm * N + cw] = v;
      }
    }
  }
}

// implicit-GEMM 5x5 conv, Cin=Cout=64, SAME, fused BN(scale/shift)+ReLU, bf16 out.
// One wave: 16 pixels x 16 couts; K-loop = 25 taps x 2 cin-chunks of 32.
__global__ void conv64_wmma_kernel(const unsigned short* __restrict__ in,   // [H*W*64]
                                   const unsigned short* __restrict__ wT,   // [25*64*64] (tap,co,ci)
                                   const float* __restrict__ scale,
                                   const float* __restrict__ shift,
                                   unsigned short* __restrict__ out,        // [H*W*64]
                                   int Himg, int Wimg) {
  const int C = 64;
  int npix = Himg * Wimg;
  int p0 = blockIdx.x << 4;
  int n0 = blockIdx.y << 4;
  int lane = threadIdx.x;
  int half = lane >> 4, q = lane & 15;
  int pix = p0 + q;
  int py = pix / Wimg;
  int px = pix - py * Wimg;
  bool prow = pix < npix;
  v8f acc;
#pragma unroll
  for (int r = 0; r < 8; ++r) acc[r] = 0.f;
  const unsigned short* bpb = wT + (size_t)(n0 + q) * C;
  for (int tap = 0; tap < 25; ++tap) {
    int iy = py + tap / 5 - 2;
    int ix = px + tap % 5 - 2;
    bool ok = prow && iy >= 0 && iy < Himg && ix >= 0 && ix < Wimg;
    const unsigned short* ap = in;
    if (ok) ap = in + (size_t)(iy * Wimg + ix) * C;
    const unsigned short* bp = bpb + (size_t)tap * C * C;
#pragma unroll
    for (int kc = 0; kc < 64; kc += 32) {
      int kb = kc + half * 8;
      v8us alo, ahi;
      if (ok) {
        alo = *(const v8us*)(ap + kb);
        ahi = *(const v8us*)(ap + kb + 16);
      } else {
#pragma unroll
        for (int j = 0; j < 8; ++j) { alo[j] = 0; ahi[j] = 0; }
      }
      v8us blo = *(const v8us*)(bp + kb);
      v8us bhi = *(const v8us*)(bp + kb + 16);
      v16us au, bu;
#pragma unroll
      for (int j = 0; j < 8; ++j) {
        au[j] = alo[j]; au[j + 8] = ahi[j];
        bu[j] = blo[j]; bu[j + 8] = bhi[j];
      }
      acc = __builtin_amdgcn_wmma_f32_16x16x32_bf16(
          false, __builtin_bit_cast(v16bf, au),
          false, __builtin_bit_cast(v16bf, bu),
          (short)0, acc, false, false);
    }
  }
  int co = n0 + q;
  float sc = scale[co], sh = shift[co];
#pragma unroll
  for (int r = 0; r < 8; ++r) {
    int pm = p0 + half * 8 + r;
    if (pm < npix) {
      float v = sc * acc[r] + sh;
      v = v > 0.f ? v : 0.f;
      out[(size_t)pm * C + co] = f2bf(v);
    }
  }
}

// ---------------- small direct convs (Cin=1 / Cout=1) ----------------

__global__ void conv0_kernel(const float* __restrict__ in,    // [H*W]
                             const float* __restrict__ w0,    // [25*64] (tap,co)
                             const float* __restrict__ b0,
                             unsigned short* __restrict__ out,  // [H*W*64] bf16 relu
                             int Himg, int Wimg) {
  int t = blockIdx.x * blockDim.x + threadIdx.x;
  int npix = Himg * Wimg;
  if (t >= npix * 64) return;
  int co = t & 63;
  int pix = t >> 6;
  int py = pix / Wimg, px = pix - py * Wimg;
  float s = b0[co];
  for (int tap = 0; tap < 25; ++tap) {
    int iy = py + tap / 5 - 2, ix = px + tap % 5 - 2;
    if (iy >= 0 && iy < Himg && ix >= 0 && ix < Wimg)
      s += in[iy * Wimg + ix] * w0[tap * 64 + co];
  }
  s = s > 0.f ? s : 0.f;
  out[t] = f2bf(s);
}

__global__ void convend_kernel(const unsigned short* __restrict__ in,  // [H*W*64] bf16
                               const float* __restrict__ wend,         // [25*64] (tap,ci)
                               const float* __restrict__ bend,
                               const float* __restrict__ skip,
                               float* __restrict__ out, int Himg, int Wimg) {
  int pix = blockIdx.x * blockDim.x + threadIdx.x;
  int npix = Himg * Wimg;
  if (pix >= npix) return;
  int py = pix / Wimg, px = pix - py * Wimg;
  float s = bend[0];
  for (int tap = 0; tap < 25; ++tap) {
    int iy = py + tap / 5 - 2, ix = px + tap % 5 - 2;
    if (iy < 0 || iy >= Himg || ix < 0 || ix >= Wimg) continue;
    const unsigned short* ip = in + (size_t)(iy * Wimg + ix) * 64;
    const float* wp = wend + tap * 64;
#pragma unroll 8
    for (int c = 0; c < 64; ++c) s += bf2f(ip[c]) * wp[c];
  }
  out[pix] = s + skip[pix];
}

// ---------------- host orchestration ----------------

struct BP {  // prepared per-branch params
  const float *b0, *bend, *w0, *wend;
  unsigned short *w1T, *w2T;
  float *sc1, *sh1, *sc2, *sh2;
};

static void run_cnn(const float* in_f32, int Himg, int Wimg, const BP& p,
                    const float* skip, float* out,
                    unsigned short* actA, unsigned short* actB, hipStream_t s) {
  int npix = Himg * Wimg;
  conv0_kernel<<<CDIV(npix * 64, 256), 256, 0, s>>>(in_f32, p.w0, p.b0, actA, Himg, Wimg);
  dim3 gc(CDIV(npix, 16), 4);
  conv64_wmma_kernel<<<gc, 32, 0, s>>>(actA, p.w1T, p.sc1, p.sh1, actB, Himg, Wimg);
  conv64_wmma_kernel<<<gc, 32, 0, s>>>(actB, p.w2T, p.sc2, p.sh2, actA, Himg, Wimg);
  convend_kernel<<<CDIV(npix, 256), 256, 0, s>>>(actA, p.wend, p.bend, skip, out, Himg, Wimg);
}

// all pipeline GEMMs are [360 x 361] = A(pad 368x384) @ Bt(pad 368x384)^T
static void gemm(const unsigned short* A, const unsigned short* Bt, float* C,
                 float alpha, int acc, hipStream_t s) {
  dim3 g(kRp / 16, kRp / 16);  // 23 x 23 tiles
  gemm_bf16_wmma_kernel<<<g, 32, 0, s>>>(A, Bt, C, kNAng, kLA, alpha, acc);
}

static void run_one_iter(const float* fin, const int* A_idx, const float* A_val,
                         const BP& ct, const BP& sb,
                         const unsigned short* cosB, const unsigned short* sinB,
                         float* sinbuf, unsigned short* sinbfP,
                         float* Fr, float* Fi, float* zR, float* zI,
                         unsigned short* zRbfP, unsigned short* zIbfP,
                         unsigned short* actA, unsigned short* actB,
                         float* z_out, float* u_out, hipStream_t s) {
  spmv_kernel<<<CDIV(kM, 256), 256, 0, s>>>(A_idx, A_val, fin, sinbuf, kM, 32);
  cvt_pad_kernel<<<CDIV(kPadElems, 256), 256, 0, s>>>(sinbuf, sinbfP);
  gemm(sinbfP, cosB, Fr, 1.f, 0, s);                           // F.real
  gemm(sinbfP, sinB, Fi, -1.f, 0, s);                          // F.imag
  run_cnn(Fr, kNAng, kLA, sb, Fr, zR, actA, actB, s);          // zR = Fr + cnn(Fr)
  run_cnn(Fi, kNAng, kLA, sb, Fi, zI, actA, actB, s);          // zI = Fi + cnn(Fi)
  cvt_pad_kernel<<<CDIV(kPadElems, 256), 256, 0, s>>>(zR, zRbfP);
  cvt_pad_kernel<<<CDIV(kPadElems, 256), 256, 0, s>>>(zI, zIbfP);
  const float inv = 1.f / (float)kLA;
  gemm(zRbfP, cosB, z_out, inv, 0, s);                         // Re(ifft)
  gemm(zIbfP, sinB, z_out, -inv, 1, s);
  run_cnn(fin, kH, kH, ct, fin, u_out, actA, actB, s);         // u = f + cnn(f)
}

extern "C" void kernel_launch(void* const* d_in, const int* in_sizes, int n_in,
                              void* d_out, int out_size, void* d_ws, size_t ws_size,
                              hipStream_t stream) {
  (void)in_sizes; (void)n_in; (void)out_size; (void)ws_size;
  const float* g      = (const float*)d_in[0];
  const float* f0     = (const float*)d_in[1];
  const int*   A_idx  = (const int*)d_in[2];
  const float* A_val  = (const float*)d_in[3];
  const int*   AT_idx = (const int*)d_in[4];
  const float* AT_val = (const float*)d_in[5];
  const int*   Ac_idx = (const int*)d_in[6];
  const float* Ac_val = (const float*)d_in[7];
  const float* w_c    = (const float*)d_in[8];
  const float* alpha  = (const float*)d_in[9];
  const float* l1 = (const float*)d_in[66];
  const float* l2 = (const float*)d_in[67];
  const float* l3 = (const float*)d_in[68];
  const float* l4 = (const float*)d_in[69];

  size_t off = 0;
  auto alloc = [&](size_t bytes) -> void* {
    off = (off + 255) & ~(size_t)255;
    void* p = (char*)d_ws + off;
    off += bytes;
    return p;
  };

  // --- per-branch weight prep (iters[0].ct, iters[0].sin, iters[1].ct, iters[1].sin) ---
  const int bases[4] = {10, 24, 38, 52};
  BP bp[4];
  for (int i = 0; i < 4; ++i) {
    int b = bases[i];
    bp[i].b0   = (const float*)d_in[b + 0];
    bp[i].bend = (const float*)d_in[b + 1];
    const float* bn1b = (const float*)d_in[b + 2];
    const float* bn1g = (const float*)d_in[b + 3];
    const float* bn1m = (const float*)d_in[b + 4];
    const float* bn1v = (const float*)d_in[b + 5];
    const float* bn2b = (const float*)d_in[b + 6];
    const float* bn2g = (const float*)d_in[b + 7];
    const float* bn2m = (const float*)d_in[b + 8];
    const float* bn2v = (const float*)d_in[b + 9];
    bp[i].w0   = (const float*)d_in[b + 10];
    const float* w1 = (const float*)d_in[b + 11];
    const float* w2 = (const float*)d_in[b + 12];
    bp[i].wend = (const float*)d_in[b + 13];
    bp[i].w1T = (unsigned short*)alloc(25 * 64 * 64 * sizeof(unsigned short));
    bp[i].w2T = (unsigned short*)alloc(25 * 64 * 64 * sizeof(unsigned short));
    bp[i].sc1 = (float*)alloc(64 * 4); bp[i].sh1 = (float*)alloc(64 * 4);
    bp[i].sc2 = (float*)alloc(64 * 4); bp[i].sh2 = (float*)alloc(64 * 4);
    wtrans_kernel<<<CDIV(25 * 64 * 64, 256), 256, 0, stream>>>(w1, bp[i].w1T, 64, 64);
    wtrans_kernel<<<CDIV(25 * 64 * 64, 256), 256, 0, stream>>>(w2, bp[i].w2T, 64, 64);
    bn_prep_kernel<<<1, 64, 0, stream>>>(bn1g, bn1b, bn1m, bn1v, bp[i].sc1, bp[i].sh1);
    bn_prep_kernel<<<1, 64, 0, stream>>>(bn2g, bn2b, bn2m, bn2v, bp[i].sc2, bp[i].sh2);
  }

  // --- DFT / filter bases (padded Bt layout) ---
  unsigned short* cosB  = (unsigned short*)alloc((size_t)kPadElems * 2);
  unsigned short* sinB  = (unsigned short*)alloc((size_t)kPadElems * 2);
  unsigned short* filtB = (unsigned short*)alloc((size_t)kPadElems * 2);
  dft_basis_kernel<<<CDIV(kPadElems, 256), 256, 0, stream>>>(cosB, sinB);
  filt_basis_kernel<<<CDIV(kPadElems, 256), 256, 0, stream>>>(w_c, filtB);

  // --- work buffers ---
  float* sinbuf = (float*)alloc((size_t)kM * 4);
  unsigned short* sinbfP = (unsigned short*)alloc((size_t)kPadElems * 2);
  float* Fr = (float*)alloc((size_t)kM * 4);
  float* Fi = (float*)alloc((size_t)kM * 4);
  float* zR = (float*)alloc((size_t)kM * 4);
  float* zI = (float*)alloc((size_t)kM * 4);
  unsigned short* zRbfP = (unsigned short*)alloc((size_t)kPadElems * 2);
  unsigned short* zIbfP = (unsigned short*)alloc((size_t)kPadElems * 2);
  float* z0 = (float*)alloc((size_t)kM * 4);
  float* u0 = (float*)alloc((size_t)kNPix * 4);
  float* gC = (float*)alloc((size_t)kM * 4);
  float* a1 = (float*)alloc((size_t)kM * 4);
  float* combo = (float*)alloc((size_t)kM * 4);
  unsigned short* s1bfP = (unsigned short*)alloc((size_t)kPadElems * 2);
  float* filt = (float*)alloc((size_t)kM * 4);
  float* a3 = (float*)alloc((size_t)kNPix * 4);
  float* f1 = (float*)alloc((size_t)kNPix * 4);
  unsigned short* actA = (unsigned short*)alloc((size_t)kM * 64 * 2);
  unsigned short* actB = (unsigned short*)alloc((size_t)kM * 64 * 2);

  // ===== iteration 0: (z0, u0) = one_iter(f0, iters[0]); sinbuf keeps Af0 =====
  run_one_iter(f0, A_idx, A_val, bp[0], bp[1], cosB, sinB, sinbuf, sinbfP,
               Fr, Fi, zR, zI, zRbfP, zIbfP, actA, actB, z0, u0, stream);

  // gC = [g ; Ac @ g]
  hipMemcpyAsync(gC, g, (size_t)kCRows * 4, hipMemcpyDeviceToDevice, stream);
  spmv_kernel<<<CDIV(kCRows, 256), 256, 0, stream>>>(Ac_idx, Ac_val, g, gC + kCRows,
                                                     kCRows, 8);
  // a1 = complete((Af0 - gC)[:LB])
  sub_kernel<<<CDIV(kCRows, 256), 256, 0, stream>>>(sinbuf, gC, a1, kCRows);
  spmv_kernel<<<CDIV(kCRows, 256), 256, 0, stream>>>(Ac_idx, Ac_val, a1, a1 + kCRows,
                                                     kCRows, 8);
  // combo = l1*a1 + l2*(Af0 - z0)
  combo_kernel<<<CDIV(kM, 256), 256, 0, stream>>>(a1, sinbuf, z0, l1, l2, combo, kM);
  // iradon: filter (GEMM) then AT spmv
  s1_pad_kernel<<<CDIV(kPadElems, 256), 256, 0, stream>>>(combo, alpha, s1bfP);
  gemm(s1bfP, filtB, filt, 1.f, 0, stream);
  spmv_kernel<<<CDIV(kNPix, 256), 256, 0, stream>>>(AT_idx, AT_val, filt, a3, kNPix, 64);
  // f1 = l4*f0 - a3 + l3*u0
  updatef_kernel<<<CDIV(kNPix, 256), 256, 0, stream>>>(f0, a3, u0, l3, l4, f1, kNPix);

  // ===== iteration 1: outputs (z1, u1) straight into d_out =====
  float* zout = (float*)d_out;
  float* uout = zout + kM;
  run_one_iter(f1, A_idx, A_val, bp[2], bp[3], cosB, sinB, sinbuf, sinbfP,
               Fr, Fi, zR, zI, zRbfP, zIbfP, actA, actB, zout, uout, stream);
}
